// ContextualAttention_90443421319191
// MI455X (gfx1250) — compile-verified, matrix operand order checked by reading, not run on hardware
//
#include <hip/hip_runtime.h>
#include <hip/hip_bf16.h>
#include <math.h>

// CDNA5 / gfx1250 fused contextual attention.
// Pipeline: pool_mask -> fmask -> wprep (w patches f16 + 1/norm) ->
//           flash (score GEMM + streaming softmax/argmax + output GEMM, WMMA f16->f32,
//                  W tiles staged via GLOBAL_LOAD_ASYNC_TO_LDS_B128 / ASYNCcnt)
//           -> scatter (conv_transpose spatial gather).
// Workspace requirement: ~61 MB.

typedef __attribute__((ext_vector_type(16))) _Float16 v16h;
typedef __attribute__((ext_vector_type(8)))  float    v8f;

#define BATCH 4
#define CH    64
#define HH    64
#define WW    64
#define LTOT  4096          // number of background patches (== queries)
#define KDIM  576           // C * 3 * 3
#define QT    16            // queries per workgroup (one WMMA M tile)
#define LT    64            // patches per flash iteration
#define NCH   192           // K/N columns staged per LDS chunk
#define SCALE_F 10.0f

// 16-bit A/B operand K index for element e of the v16h, per CDNA5 ISA layout:
// lanes' VGPR0..3 hold K = half*8 + {0..7}, VGPR4..7 hold K = 16 + half*8 + {0..7}
__device__ __forceinline__ int kmap16(int e, int half) {
    int vi = e >> 1, lo = e & 1;
    int base = (vi < 4) ? (vi * 2 + lo) : (16 + (vi - 4) * 2 + lo);
    return base + half * 8;
}

// Low 32 bits of a flat shared-aperture address == wave-relative LDS byte offset.
__device__ __forceinline__ unsigned lds_off(const void* p) {
    return (unsigned)(unsigned long long)p;
}

// CDNA5 async global->LDS copy, 16B per lane, tracked with ASYNCcnt.
__device__ __forceinline__ void async_load_b128(unsigned lds_addr, const void* gptr) {
    asm volatile("global_load_async_to_lds_b128 %0, %1, off"
                 :: "v"(lds_addr), "v"((unsigned long long)gptr)
                 : "memory");
}
__device__ __forceinline__ void wait_async0() {
    asm volatile("s_wait_asynccnt 0x0" ::: "memory");
}

// ---------------- mask 4x4 average pool: [B,1,256,256] -> [B,64,64] ----------------
__global__ void pool_mask_k(const float* __restrict__ mask, float* __restrict__ mp) {
    int idx = blockIdx.x * blockDim.x + threadIdx.x;
    if (idx >= BATCH * HH * WW) return;
    int b = idx / (HH * WW);
    int r = idx % (HH * WW);
    int y = r / WW, x = r % WW;
    const float* m = mask + (size_t)b * 256 * 256;
    float s = 0.f;
#pragma unroll
    for (int dy = 0; dy < 4; ++dy)
#pragma unroll
        for (int dx = 0; dx < 4; ++dx)
            s += m[(y * 4 + dy) * 256 + (x * 4 + dx)];
    mp[idx] = s * (1.f / 16.f);
}

// ---------------- f_masked = f * (1 - mp[b]) ----------------
__global__ void fmask_k(const float* __restrict__ f, const float* __restrict__ mp,
                        float* __restrict__ fm) {
    int idx = blockIdx.x * blockDim.x + threadIdx.x;
    if (idx >= BATCH * CH * HH * WW) return;
    int b = idx / (CH * HH * WW);
    int pix = idx % (HH * WW);
    fm[idx] = f[idx] * (1.f - mp[b * HH * WW + pix]);
}

// ---------------- w patches (f16) + rnorm = 1/sqrt(sum w^2 + eps) ----------------
// w[b,l,c,i,j] = bpad[b,c,ly+i-1,lx+j-1] * m0pad[ly+i-1,lx+j-1]   (m0 from batch 0 mask)
__global__ void wprep_k(const float* __restrict__ bg, const float* __restrict__ mp,
                        _Float16* __restrict__ w_h, float* __restrict__ rnorm) {
    int bl = blockIdx.x;                 // b*4096 + l
    int l  = bl & (LTOT - 1);
    int b  = bl >> 12;
    int ly = l / WW, lx = l % WW;
    int c  = threadIdx.x;                // 0..63
    const float* bp  = bg + ((size_t)b * CH + c) * HH * WW;
    const float* mp0 = mp;               // batch 0 quirk of the reference
    _Float16* wrow = w_h + (size_t)bl * KDIM + c * 9;
    float ssum = 0.f;
#pragma unroll
    for (int i = 0; i < 3; ++i)
#pragma unroll
        for (int j = 0; j < 3; ++j) {
            int yy = ly + i - 1, xx = lx + j - 1;
            float wv = 0.f;
            if (yy >= 0 && yy < HH && xx >= 0 && xx < WW)
                wv = bp[yy * WW + xx] * mp0[yy * WW + xx];
            wrow[i * 3 + j] = (_Float16)wv;
            ssum += wv * wv;
        }
    __shared__ float red[64];
    red[c] = ssum;
    __syncthreads();
    for (int off = 32; off > 0; off >>= 1) {
        if (c < off) red[c] += red[c + off];
        __syncthreads();
    }
    if (c == 0) rnorm[bl] = rsqrtf(red[0] + 1e-4f);
}

// ---------------- fused flash attention kernel ----------------
__global__ __launch_bounds__(128) void flash_k(
    const float* __restrict__ fm, const _Float16* __restrict__ w_h,
    const float* __restrict__ rnorm, float* __restrict__ Obuf,
    float* __restrict__ out_off) {
    const int b    = blockIdx.y;
    const int q0   = blockIdx.x * QT;
    const int tid  = threadIdx.x;
    const int wv   = tid >> 5;           // wave 0..3
    const int lane = tid & 31;
    const int half = lane >> 4;
    const int nl   = lane & 15;

    __shared__ alignas(16) _Float16 Afp[QT][KDIM + 8];   // queries x K, f16
    __shared__ alignas(16) _Float16 Wc[LT][NCH + 8];     // staged W tile (padded stride)
    __shared__ float    Sl[QT][LT + 4];                  // scaled scores
    __shared__ _Float16 Pl[QT][LT + 8];                  // exp probs, f16
    __shared__ float pmax[QT][8];
    __shared__ int   parg[QT][8];
    __shared__ float psum[QT][8];
    __shared__ float run_max[QT], run_sum[QT], scl[QT], newm[QT];
    __shared__ int   run_arg[QT];

    int km[16];
#pragma unroll
    for (int e = 0; e < 16; ++e) km[e] = kmap16(e, half);

    // Stage unfolded f_masked patches for our 16 queries (zero-padded 3x3 gather).
    for (int idx = tid; idx < QT * KDIM; idx += 128) {
        int qr = idx / KDIM, k = idx % KDIM;
        int c = k / 9, ij = k % 9, i = ij / 3, j = ij % 3;
        int q = q0 + qr;
        int qy = q / WW, qx = q % WW;
        int yy = qy + i - 1, xx = qx + j - 1;
        float v = 0.f;
        if (yy >= 0 && yy < HH && xx >= 0 && xx < WW)
            v = fm[(((size_t)b * CH + c) * HH + yy) * WW + xx];
        Afp[qr][k] = (_Float16)v;
    }
    if (tid < QT) { run_max[tid] = -1e30f; run_sum[tid] = 0.f; run_arg[tid] = 0; }
    __syncthreads();

    v8f acc[9];
#pragma unroll
    for (int s9 = 0; s9 < 9; ++s9) acc[s9] = (v8f){0.f, 0.f, 0.f, 0.f, 0.f, 0.f, 0.f, 0.f};

    for (int l0 = 0; l0 < LTOT; l0 += LT) {
        // Speculative prefetch of the next L-tile of w_h (global_prefetch_b8).
        if (l0 + LT < LTOT) {
            const char* nb = (const char*)(w_h + ((size_t)(b * LTOT + l0 + LT)) * KDIM);
#pragma unroll
            for (int pf = 0; pf < 5; ++pf) {
                int off = tid * 128 + pf * 16384;
                if (off < LT * KDIM * 2) __builtin_prefetch(nb + off, 0, 3);
            }
        }

        // ---- score GEMM: S[16, 64] = Afp[16,576] * W^T, K chunked ----
        v8f sacc = (v8f){0.f, 0.f, 0.f, 0.f, 0.f, 0.f, 0.f, 0.f};
        for (int kk = 0; kk < KDIM; kk += NCH) {
            __syncthreads();
            for (int idx = tid; idx < LT * (NCH / 8); idx += 128) {
                int row = idx / (NCH / 8), kb = idx % (NCH / 8);
                const _Float16* g =
                    w_h + ((size_t)(b * LTOT + l0 + row)) * KDIM + kk + kb * 8;
                async_load_b128(lds_off(&Wc[row][kb * 8]), g);
            }
            wait_async0();
            __syncthreads();
#pragma unroll
            for (int ks = 0; ks < NCH / 32; ++ks) {
                v16h a, bb;
#pragma unroll
                for (int e = 0; e < 16; ++e) a[e]  = Afp[nl][kk + ks * 32 + km[e]];
#pragma unroll
                for (int e = 0; e < 16; ++e) bb[e] = Wc[wv * 16 + nl][ks * 32 + km[e]];
                sacc = __builtin_amdgcn_wmma_f32_16x16x32_f16(
                    false, a, false, bb, (short)0, sacc, false, false);
            }
        }
        // spill S (apply 1/norm and softmax scale); C-tile layout: row = v + 8*half, col = nl
        {
            int lcol = wv * 16 + nl;
            float rs = rnorm[b * LTOT + l0 + lcol] * SCALE_F;
#pragma unroll
            for (int v = 0; v < 8; ++v)
                Sl[v + 8 * half][lcol] = sacc[v] * rs;
        }
        __syncthreads();

        // ---- streaming softmax + running argmax ----
        {
            int row = tid & 15, seg = tid >> 4;
            float mx = -1e30f; int ar = 0;
#pragma unroll
            for (int jj = 0; jj < 8; ++jj) {
                float v = Sl[row][seg * 8 + jj];
                if (v > mx) { mx = v; ar = seg * 8 + jj; }
            }
            pmax[row][seg] = mx; parg[row][seg] = ar;
        }
        __syncthreads();
        if (tid < QT) {
            float mx = -1e30f; int ar = 0;
#pragma unroll
            for (int s = 0; s < 8; ++s) {
                float v = pmax[tid][s];
                if (v > mx) { mx = v; ar = parg[tid][s]; }
            }
            float om = run_max[tid];
            if (mx > om) run_arg[tid] = l0 + ar;
            float nm = fmaxf(om, mx);
            newm[tid]  = nm;
            scl[tid]   = __expf(om - nm);
            run_max[tid] = nm;
        }
        __syncthreads();
        {
            int row = tid & 15, seg = tid >> 4;
            float nm = newm[row];
            float s = 0.f;
#pragma unroll
            for (int jj = 0; jj < 8; ++jj) {
                float p = __expf(Sl[row][seg * 8 + jj] - nm);
                Pl[row][seg * 8 + jj] = (_Float16)p;
                s += p;
            }
            psum[row][seg] = s;
        }
        __syncthreads();
        if (tid < QT) {
            float s = run_sum[tid] * scl[tid];
#pragma unroll
            for (int g = 0; g < 8; ++g) s += psum[tid][g];
            run_sum[tid] = s;
        }
        // rescale output accumulators by exp(old_max - new_max)
        float sc[8];
#pragma unroll
        for (int v = 0; v < 8; ++v) sc[v] = scl[v + 8 * half];
#pragma unroll
        for (int s9 = 0; s9 < 9; ++s9)
#pragma unroll
            for (int v = 0; v < 8; ++v) acc[s9][v] *= sc[v];

        // ---- output GEMM: O[16, 576] += P[16,64] * W[64,576], N chunked ----
        for (int chn = 0; chn < 3; ++chn) {
            int nn = chn * NCH;
            __syncthreads();
            for (int idx = tid; idx < LT * (NCH / 8); idx += 128) {
                int row = idx / (NCH / 8), kb = idx % (NCH / 8);
                const _Float16* g =
                    w_h + ((size_t)(b * LTOT + l0 + row)) * KDIM + nn + kb * 8;
                async_load_b128(lds_off(&Wc[row][kb * 8]), g);
            }
            wait_async0();
            __syncthreads();
            v16h a2[2];
#pragma unroll
            for (int ksp = 0; ksp < 2; ++ksp)
#pragma unroll
                for (int e = 0; e < 16; ++e) a2[ksp][e] = Pl[nl][ksp * 32 + km[e]];
#pragma unroll
            for (int s3 = 0; s3 < 3; ++s3) {
                int s9    = chn * 3 + s3;
                int nbase = (wv + 4 * s9) * 16 - nn;   // wave owns n-tiles wv, wv+4, ...
#pragma unroll
                for (int ksp = 0; ksp < 2; ++ksp) {
                    v16h bb;
#pragma unroll
                    for (int e = 0; e < 16; ++e) bb[e] = Wc[ksp * 32 + km[e]][nbase + nl];
                    acc[s9] = __builtin_amdgcn_wmma_f32_16x16x32_f16(
                        false, a2[ksp], false, bb, (short)0, acc[s9], false, false);
                }
            }
        }
    }

    // ---- finalize: divide by softmax denominator, write O and argmax ----
    __syncthreads();
    float inv[8];
#pragma unroll
    for (int v = 0; v < 8; ++v) inv[v] = 1.f / run_sum[v + 8 * half];
#pragma unroll
    for (int s9 = 0; s9 < 9; ++s9) {
        int n = (wv + 4 * s9) * 16 + nl;
#pragma unroll
        for (int v = 0; v < 8; ++v) {
            int m = v + 8 * half;
            Obuf[((size_t)(b * LTOT + q0 + m)) * KDIM + n] = acc[s9][v] * inv[v];
        }
    }
    if (tid < QT) out_off[b * LTOT + q0 + tid] = (float)run_arg[tid];
}

// ---------------- conv_transpose epilogue: gather 3x3 neighborhood of O ----------------
__global__ void scatter_k(const float* __restrict__ Obuf, float* __restrict__ y) {
    int idx = blockIdx.x * blockDim.x + threadIdx.x;
    if (idx >= BATCH * CH * HH * WW) return;
    int b = idx / (CH * HH * WW);
    int r = idx % (CH * HH * WW);
    int c = r / (HH * WW);
    int pr = r % (HH * WW);
    int py = pr / WW, px = pr % WW;
    float s = 0.f;
#pragma unroll
    for (int i = 0; i < 3; ++i) {
        int qy = py + 1 - i;
        if (qy < 0 || qy >= HH) continue;
#pragma unroll
        for (int j = 0; j < 3; ++j) {
            int qx = px + 1 - j;
            if (qx < 0 || qx >= WW) continue;
            int q = qy * WW + qx;
            s += Obuf[((size_t)(b * LTOT + q)) * KDIM + c * 9 + i * 3 + j];
        }
    }
    y[idx] = s;
}

extern "C" void kernel_launch(void* const* d_in, const int* in_sizes, int n_in,
                              void* d_out, int out_size, void* d_ws, size_t ws_size,
                              hipStream_t stream) {
    const float* bg   = (const float*)d_in[0];   // [4,64,64,64]
    const float* f    = (const float*)d_in[1];   // [4,64,64,64]
    const float* mask = (const float*)d_in[2];   // [4,1,256,256]
    float* y_out   = (float*)d_out;                       // [4,64,64,64]
    float* off_out = (float*)d_out + BATCH * CH * HH * WW; // [4,1,64,64] argmax

    // Workspace layout (bytes), total ~61 MB.
    char* ws = (char*)d_ws;
    float*    mp   = (float*)(ws + 0);           // 64 KB
    float*    rnm  = (float*)(ws + 65536);       // 64 KB
    float*    fmk  = (float*)(ws + 131072);      // 4 MB
    _Float16* wh   = (_Float16*)(ws + 4325376);  // 18.9 MB
    float*    Obuf = (float*)(ws + 23199744);    // 37.7 MB

    pool_mask_k<<<dim3(BATCH * HH * WW / 256), 256, 0, stream>>>(mask, mp);
    fmask_k<<<dim3(BATCH * CH * HH * WW / 256), 256, 0, stream>>>(f, mp, fmk);
    wprep_k<<<dim3(BATCH * LTOT), 64, 0, stream>>>(bg, mp, wh, rnm);
    flash_k<<<dim3(LTOT / QT, BATCH), 128, 0, stream>>>(fmk, wh, rnm, Obuf, off_out);
    scatter_k<<<dim3(BATCH * CH * HH * WW / 256), 256, 0, stream>>>(Obuf, y_out);
}
